// PointDistanceAttention_2619930051078
// MI455X (gfx1250) — compile-verified, matrix operand order checked by reading, not run on hardware
//
#include <hip/hip_runtime.h>
#include <hip/hip_bf16.h>

// ---------------------------------------------------------------------------
// Types for CDNA5 WMMA
// ---------------------------------------------------------------------------
typedef __attribute__((ext_vector_type(16))) __bf16    v16bf;
typedef __attribute__((ext_vector_type(8)))  float     v8f;
typedef __attribute__((ext_vector_type(8)))  unsigned  v8u;

#define N_FREQ   4
#define KPTS     8        // K points per ray
#define NDIR     6        // feat map dirs
#define NROWS    48       // KPTS*NDIR
#define W_POSE   256
#define W_RAY    64
#define KQ_LEN   16
#define V_LEN    128
#define IN_POSE_PAD 64    // 51 padded to 64 (2 bf16 k-steps)
#define IN_RAY_PAD  32    // 27 padded to 32 (1 bf16 k-step)

// f32 -> bf16 bits via native conversion (RNE, lowers to v_cvt_pk_bf16_f32)
__device__ __forceinline__ unsigned short f2bf(float f) {
    __bf16 h = (__bf16)f;
    return __builtin_bit_cast(unsigned short, h);
}

// 32 contiguous bytes (16 bf16) -> v16bf  (2x b128 loads)
__device__ __forceinline__ v16bf ld_bf16x16(const unsigned short* p) {
    const uint4* q = reinterpret_cast<const uint4*>(p);
    uint4 a = q[0], b = q[1];
    v8u u = { a.x, a.y, a.z, a.w, b.x, b.y, b.z, b.w };
    return __builtin_bit_cast(v16bf, u);
}

// ---------------------------------------------------------------------------
// Kernel 1: weight transpose + f32->bf16.  All Wt layouts are [N][K] so a
// WMMA B lane (n = lane&15, k-half = lane>>4) reads 32 contiguous bytes.
// ---------------------------------------------------------------------------
__global__ void pda_prep_weights(const float* __restrict__ W1,   // [51,256]
                                 const float* __restrict__ W2,   // [256,256]
                                 const float* __restrict__ Wk,   // [256,16]
                                 const float* __restrict__ Wr,   // [27,64]
                                 const float* __restrict__ Wq,   // [64,16]
                                 unsigned short* __restrict__ W1t,  // [256][64]
                                 unsigned short* __restrict__ W2t,  // [256][256]
                                 unsigned short* __restrict__ Wkt,  // [16][256]
                                 unsigned short* __restrict__ Wrt,  // [64][32]
                                 unsigned short* __restrict__ Wqt)  // [16][64]
{
    int i = blockIdx.x * blockDim.x + threadIdx.x;          // 65536 threads
    if (i < 256 * 64) {
        int n = i >> 6, k = i & 63;
        W1t[i] = (k < 51) ? f2bf(W1[k * 256 + n]) : (unsigned short)0;
    }
    if (i < 256 * 256) {
        int n = i >> 8, k = i & 255;
        W2t[i] = f2bf(W2[k * 256 + n]);
    }
    if (i < 16 * 256) {
        int n = i >> 8, k = i & 255;
        Wkt[i] = f2bf(Wk[k * 16 + n]);
    }
    if (i < 64 * 32) {
        int n = i >> 5, k = i & 31;
        Wrt[i] = (k < 27) ? f2bf(Wr[k * 64 + n]) : (unsigned short)0;
    }
    if (i < 16 * 64) {
        int n = i >> 6, k = i & 63;
        Wqt[i] = f2bf(Wq[k * 16 + n]);
    }
}

// ---------------------------------------------------------------------------
// Kernel 2: WMMA query MLP. One wave handles 16 rays:
//   A = posenc(dirs)[16,32] built directly in registers in this kernel's
//   contiguous k order (lane&15 = ray/row, lane>>4 = K half: k = 16*half + s), B = Wrt/Wqt bf16 [N][K].
// ---------------------------------------------------------------------------
__global__ __launch_bounds__(256)
void pda_query(const float* __restrict__ dirs,           // [BR,3]
               const unsigned short* __restrict__ Wrt,   // [64][32] bf16
               const float* __restrict__ br,             // [64]
               const unsigned short* __restrict__ Wqt,   // [16][64] bf16
               const float* __restrict__ bq,             // [16]
               float* __restrict__ qout,                 // [BR,16]
               int nrays)
{
    __shared__ unsigned short sH[8 * 16 * W_RAY];        // 16 KB: h[16][64]/wave

    const int tid  = threadIdx.x;
    const int wave = tid >> 5;
    const int lane = tid & 31;
    const int l15  = lane & 15;
    const int half = lane >> 4;
    const int rayBase = blockIdx.x * 128 + wave * 16;

    int ray = rayBase + l15;
    if (ray >= nrays) ray = nrays - 1;                   // clamp (safe dup)
    float d0 = dirs[(size_t)ray * 3 + 0];
    float d1 = dirs[(size_t)ray * 3 + 1];
    float d2 = dirs[(size_t)ray * 3 + 2];

    // posenc order: [d0,d1,d2, f: sin(xyz),cos(xyz) for f=1,2,4,8], pad to 32
    v16bf a;
    if (half == 0) {            // K = 0..15
        a[0]  = (__bf16)d0;            a[1]  = (__bf16)d1;
        a[2]  = (__bf16)d2;
        a[3]  = (__bf16)__sinf(d0);    a[4]  = (__bf16)__sinf(d1);
        a[5]  = (__bf16)__sinf(d2);
        a[6]  = (__bf16)__cosf(d0);    a[7]  = (__bf16)__cosf(d1);
        a[8]  = (__bf16)__cosf(d2);
        a[9]  = (__bf16)__sinf(2.f*d0); a[10] = (__bf16)__sinf(2.f*d1);
        a[11] = (__bf16)__sinf(2.f*d2);
        a[12] = (__bf16)__cosf(2.f*d0); a[13] = (__bf16)__cosf(2.f*d1);
        a[14] = (__bf16)__cosf(2.f*d2);
        a[15] = (__bf16)__sinf(4.f*d0);
    } else {                    // K = 16..31
        a[0]  = (__bf16)__sinf(4.f*d1); a[1]  = (__bf16)__sinf(4.f*d2);
        a[2]  = (__bf16)__cosf(4.f*d0); a[3]  = (__bf16)__cosf(4.f*d1);
        a[4]  = (__bf16)__cosf(4.f*d2);
        a[5]  = (__bf16)__sinf(8.f*d0); a[6]  = (__bf16)__sinf(8.f*d1);
        a[7]  = (__bf16)__sinf(8.f*d2);
        a[8]  = (__bf16)__cosf(8.f*d0); a[9]  = (__bf16)__cosf(8.f*d1);
        a[10] = (__bf16)__cosf(8.f*d2);
        a[11] = (__bf16)0.f; a[12] = (__bf16)0.f; a[13] = (__bf16)0.f;
        a[14] = (__bf16)0.f; a[15] = (__bf16)0.f;
    }

    // layer: h[16,64] = pe @ Wr + br   (no activation, D=1)
    unsigned short* hRow = &sH[wave * 16 * W_RAY];
#pragma unroll
    for (int nt = 0; nt < 4; ++nt) {
        v16bf b = ld_bf16x16(Wrt + (nt * 16 + l15) * IN_RAY_PAD + half * 16);
        v8f c = {0.f, 0.f, 0.f, 0.f, 0.f, 0.f, 0.f, 0.f};
        c = __builtin_amdgcn_wmma_f32_16x16x32_bf16(
                false, a, false, b, (short)0, c, false, false);
        float bv = br[nt * 16 + l15];
#pragma unroll
        for (int i = 0; i < 8; ++i)
            hRow[(i + 8 * half) * W_RAY + nt * 16 + l15] = f2bf(c[i] + bv);
    }
    __syncthreads();

    // head: q[16,16] = h @ Wq + bq
    v8f c = {0.f, 0.f, 0.f, 0.f, 0.f, 0.f, 0.f, 0.f};
#pragma unroll
    for (int ks = 0; ks < 2; ++ks) {
        v16bf ah = ld_bf16x16(hRow + l15 * W_RAY + ks * 32 + half * 16);
        v16bf b  = ld_bf16x16(Wqt + l15 * W_RAY + ks * 32 + half * 16);
        c = __builtin_amdgcn_wmma_f32_16x16x32_bf16(
                false, ah, false, b, (short)0, c, false, false);
    }
    float bv = bq[l15];
#pragma unroll
    for (int i = 0; i < 8; ++i) {
        int orow = rayBase + 8 * half + i;
        if (orow < nrays)
            qout[(size_t)orow * KQ_LEN + l15] = c[i] + bv;
    }
}

// ---------------------------------------------------------------------------
// Main-GEMM helper: each wave owns N-tiles {2w, 2w+1} for all 3 M-tiles.
// Two accumulator chains share one A operand -> independent WMMAs the
// scheduler can interleave (hides the bf16 WMMA->WMMA hazard slots).
// ---------------------------------------------------------------------------
__device__ __forceinline__ void gemm_layer(const unsigned short* __restrict__ sA, int lda,
                                           const unsigned short* __restrict__ gBt, int ldb,
                                           const float* __restrict__ bias,
                                           unsigned short* __restrict__ sOut,
                                           int ksteps, int wave, int lane, bool relu)
{
    const int half = lane >> 4;
    const int l15  = lane & 15;
    const int nt0  = wave * 2, nt1 = wave * 2 + 1;
    const unsigned short* b0row = gBt + (nt0 * 16 + l15) * ldb + half * 16;
    const unsigned short* b1row = b0row + 16 * ldb;
    const float bv0 = bias[nt0 * 16 + l15];
    const float bv1 = bias[nt1 * 16 + l15];
#pragma unroll
    for (int mt = 0; mt < 3; ++mt) {
        v8f c0 = {0.f, 0.f, 0.f, 0.f, 0.f, 0.f, 0.f, 0.f};
        v8f c1 = {0.f, 0.f, 0.f, 0.f, 0.f, 0.f, 0.f, 0.f};
        const unsigned short* arow = sA + (mt * 16 + l15) * lda + half * 16;
        for (int ks = 0; ks < ksteps; ++ks) {
            v16bf av = ld_bf16x16(arow  + ks * 32);
            v16bf b0 = ld_bf16x16(b0row + ks * 32);
            v16bf b1 = ld_bf16x16(b1row + ks * 32);
            c0 = __builtin_amdgcn_wmma_f32_16x16x32_bf16(
                     false, av, false, b0, (short)0, c0, false, false);
            c1 = __builtin_amdgcn_wmma_f32_16x16x32_bf16(
                     false, av, false, b1, (short)0, c1, false, false);
        }
#pragma unroll
        for (int i = 0; i < 8; ++i) {
            int m = mt * 16 + half * 8 + i;          // C layout: VGPR i -> M
            float v0 = c0[i] + bv0, v1 = c1[i] + bv1;
            if (relu) { v0 = fmaxf(v0, 0.f); v1 = fmaxf(v1, 0.f); }
            sOut[m * W_POSE + nt0 * 16 + l15] = f2bf(v0);
            sOut[m * W_POSE + nt1 * 16 + l15] = f2bf(v1);
        }
    }
}

// ---------------------------------------------------------------------------
// Kernel 3: main attention kernel. One block (8 wave32s) per ray.
// The 24.5 KB value tensor streams into LDS via GLOBAL_LOAD_ASYNC_TO_LDS_B128
// (ASYNCcnt) concurrently with the WMMA MLP; s_wait_asynccnt + barrier before
// the reduction consumes it.
// ---------------------------------------------------------------------------
__global__ __launch_bounds__(256)
void pda_main(const float* __restrict__ feats,       // [BR,48,128]
              const float* __restrict__ dist,        // [BR,8]
              const float* __restrict__ proj,        // [BR,8]
              const float* __restrict__ ang,         // [BR,8]
              const float* __restrict__ b1,          // [256]
              const float* __restrict__ b2,          // [256]
              const float* __restrict__ bk,          // [16]
              const unsigned short* __restrict__ W1t,// [256][64]  bf16
              const unsigned short* __restrict__ W2t,// [256][256] bf16
              const unsigned short* __restrict__ Wkt,// [16][256]  bf16
              const float* __restrict__ query,       // [BR,16]
              float* __restrict__ out)               // [BR,128]
{
    __shared__ unsigned short sX [NROWS * IN_POSE_PAD];  // 6 KB   bf16 input
    __shared__ unsigned short sH1[NROWS * W_POSE];       // 24 KB  bf16
    __shared__ unsigned short sH2[NROWS * W_POSE];       // 24 KB  bf16
    __shared__ float sFeat[NROWS * V_LEN];               // 24 KB  async values
    __shared__ float sKeys[NROWS * KQ_LEN];              // 3 KB
    __shared__ float sQ[KQ_LEN];
    __shared__ float sScore[NROWS];
    __shared__ float sAttn[NROWS];
    __shared__ float sPartV[8 * V_LEN];                  // 4 KB

    const int ray  = blockIdx.x;
    const int tid  = threadIdx.x;
    const int wave = tid >> 5;
    const int lane = tid & 31;

    // ---- kick off async copy of this ray's values into LDS (ASYNCcnt) ----
    {
        const char* gbase = (const char*)(feats + (size_t)ray * NROWS * V_LEN);
        // generic shared pointer: low 32 bits are the LDS byte address
        unsigned ldsBase = (unsigned)(unsigned long long)(uintptr_t)&sFeat[0];
#pragma unroll
        for (int r = 0; r < 6; ++r) {                    // 6 * 256 * 16B = 24 KB
            unsigned off = (unsigned)((r * 256 + tid) * 16);
            unsigned ldsAddr = ldsBase + off;
            unsigned long long ga = (unsigned long long)(uintptr_t)(gbase + off);
            asm volatile("global_load_async_to_lds_b128 %0, %1, off"
                         :: "v"(ldsAddr), "v"(ga) : "memory");
        }
    }

    // keep the L2-resident weights warm (global_prefetch_b8)
    __builtin_prefetch(W2t + (size_t)tid * 256, 0, 3);

    // ---------------- Phase 1: positional encodings into sX ---------------
    if (tid < KQ_LEN) sQ[tid] = query[(size_t)ray * KQ_LEN + tid];
    if (tid < NROWS) {
        int p = tid / NDIR, dd = tid % NDIR;
        const float* dR = dist + (size_t)ray * KPTS;
        const float* pR = proj + (size_t)ray * KPTS;
        float dmax = dR[0], pmax = pR[0];
#pragma unroll
        for (int k = 1; k < KPTS; ++k) {
            dmax = fmaxf(dmax, dR[k]);
            pmax = fmaxf(pmax, pR[k]);
        }
        float vals[3];
        vals[0] = dR[p] / dmax;
        vals[1] = pR[p] / pmax;
        vals[2] = ang[(size_t)ray * KPTS + p];
        unsigned short* row = &sX[tid * IN_POSE_PAD];
#pragma unroll
        for (int s = 0; s < 3; ++s) {                    // 3x posenc(scalar)=9
            float x = vals[s];
            row[s * 9 + 0] = f2bf(x);
#pragma unroll
            for (int f = 0; f < N_FREQ; ++f) {
                float sc = (float)(1 << f) * x;
                row[s * 9 + 1 + 2 * f] = f2bf(__sinf(sc));
                row[s * 9 + 2 + 2 * f] = f2bf(__cosf(sc));
            }
        }
        // feat-map-dir encoding (24 vals): per freq [sin xyz, cos xyz]
#pragma unroll
        for (int f = 0; f < N_FREQ; ++f) {
            float sc = (float)(1 << f);
#pragma unroll
            for (int cI = 0; cI < 3; ++cI) {
                float v = (cI == (dd >> 1)) ? ((dd & 1) ? 1.f : -1.f) : 0.f;
                row[27 + f * 6 + cI]     = f2bf(__sinf(sc * v));
                row[27 + f * 6 + 3 + cI] = f2bf(__cosf(sc * v));
            }
        }
#pragma unroll
        for (int k = 51; k < IN_POSE_PAD; ++k) row[k] = 0;
    }
    __syncthreads();

    // ---------------- Phase 2: layer1  x[48,64] @ W1t -> H1 (relu) ---------
    gemm_layer(sX, IN_POSE_PAD, W1t, IN_POSE_PAD, b1, sH1, 2, wave, lane, true);
    __syncthreads();

    // ---------------- Phase 3: layer2  H1 @ W2t -> H2 (relu) ---------------
    gemm_layer(sH1, W_POSE, W2t, W_POSE, b2, sH2, 8, wave, lane, true);
    __syncthreads();

    // ---------------- Phase 4: key head  H2 @ Wkt -> keys[48,16] -----------
    if (wave < 3) {
        int half = lane >> 4, l15 = lane & 15;
        int mt = wave;
        v8f c = {0.f, 0.f, 0.f, 0.f, 0.f, 0.f, 0.f, 0.f};
        const unsigned short* arow = sH2 + (mt * 16 + l15) * W_POSE + half * 16;
        const unsigned short* brow = Wkt + l15 * W_POSE + half * 16;
        for (int ks = 0; ks < 8; ++ks) {
            v16bf a = ld_bf16x16(arow + ks * 32);
            v16bf b = ld_bf16x16(brow + ks * 32);
            c = __builtin_amdgcn_wmma_f32_16x16x32_bf16(
                    false, a, false, b, (short)0, c, false, false);
        }
        float bv = bk[l15];
#pragma unroll
        for (int i = 0; i < 8; ++i) {
            int m = mt * 16 + half * 8 + i;
            sKeys[m * KQ_LEN + l15] = c[i] + bv;
        }
    }
    __syncthreads();

    // ---------------- Phase 5: scores + softmax ----------------------------
    if (tid < NROWS) {
        float s = 0.f;
#pragma unroll
        for (int n = 0; n < KQ_LEN; ++n) s += sKeys[tid * KQ_LEN + n] * sQ[n];
        sScore[tid] = s;
    }
    __syncthreads();
    if (tid == 0) {
        float mx = sScore[0];
        for (int j = 1; j < NROWS; ++j) mx = fmaxf(mx, sScore[j]);
        float sm = 0.f;
        for (int j = 0; j < NROWS; ++j) {
            float e = __expf(sScore[j] - mx);
            sAttn[j] = e;
            sm += e;
        }
        float inv = 1.f / sm;
        for (int j = 0; j < NROWS; ++j) sAttn[j] *= inv;
    }

    // every wave waits for its own async copies, barrier covers the rest
    asm volatile("s_wait_asynccnt 0x0" ::: "memory");
    __syncthreads();

    // ---------------- Phase 6: value reduction (from LDS) ------------------
    {
        int seg  = tid >> 5;                  // 8 segments of 6 rows
        int quad = tid & 31;                  // 32 x float4 = 128 cols
        float4 acc = make_float4(0.f, 0.f, 0.f, 0.f);
        for (int j = seg * 6; j < seg * 6 + 6; ++j) {
            float w = sAttn[j];
            float4 f4 = *reinterpret_cast<const float4*>(&sFeat[j * V_LEN + quad * 4]);
            acc.x += w * f4.x; acc.y += w * f4.y;
            acc.z += w * f4.z; acc.w += w * f4.w;
        }
        *reinterpret_cast<float4*>(&sPartV[seg * V_LEN + quad * 4]) = acc;
    }
    __syncthreads();
    if (tid < V_LEN) {
        float s = 0.f;
#pragma unroll
        for (int g = 0; g < 8; ++g) s += sPartV[g * V_LEN + tid];
        out[(size_t)ray * V_LEN + tid] = s;
    }
}

// ---------------------------------------------------------------------------
// Launcher
// ---------------------------------------------------------------------------
extern "C" void kernel_launch(void* const* d_in, const int* in_sizes, int n_in,
                              void* d_out, int out_size, void* d_ws, size_t ws_size,
                              hipStream_t stream) {
    const float* dirs  = (const float*)d_in[0];
    const float* feats = (const float*)d_in[1];
    const float* dist  = (const float*)d_in[2];
    const float* proj  = (const float*)d_in[3];
    const float* ang   = (const float*)d_in[4];
    const float* W1    = (const float*)d_in[5];
    const float* b1    = (const float*)d_in[6];
    const float* W2    = (const float*)d_in[7];
    const float* b2    = (const float*)d_in[8];
    const float* Wk    = (const float*)d_in[9];
    const float* bk    = (const float*)d_in[10];
    const float* Wr    = (const float*)d_in[11];
    const float* br    = (const float*)d_in[12];
    const float* Wq    = (const float*)d_in[13];
    const float* bq    = (const float*)d_in[14];
    float* out = (float*)d_out;

    const int nrays = in_sizes[0] / 3;            // B*R

    // scratch layout
    char* ws = (char*)d_ws;
    float* qbuf = (float*)ws;                                   // [nrays,16]
    size_t off = (size_t)nrays * KQ_LEN * sizeof(float);
    unsigned short* W1t = (unsigned short*)(ws + off);          // [256][64]
    unsigned short* W2t = W1t + 256 * IN_POSE_PAD;              // [256][256]
    unsigned short* Wkt = W2t + 256 * 256;                      // [16][256]
    unsigned short* Wrt = Wkt + 16 * 256;                       // [64][32]
    unsigned short* Wqt = Wrt + 64 * IN_RAY_PAD;                // [16][64]

    pda_prep_weights<<<256, 256, 0, stream>>>(W1, W2, Wk, Wr, Wq,
                                              W1t, W2t, Wkt, Wrt, Wqt);
    pda_query<<<(nrays + 127) / 128, 256, 0, stream>>>(dirs, Wrt, br, Wqt, bq,
                                                       qbuf, nrays);
    pda_main<<<nrays, 256, 0, stream>>>(feats, dist, proj, ang, b1, b2, bk,
                                        W1t, W2t, Wkt, qbuf, out);
}
